// GCN_88021059764775
// MI455X (gfx1250) — matrix operand order repaired and checked by reading, not compiled, not run on hardware
//
#include <hip/hip_runtime.h>
#include <hip/hip_bf16.h>

typedef _Float16 h4v  __attribute__((ext_vector_type(4)));
typedef _Float16 h8v  __attribute__((ext_vector_type(8)));
typedef _Float16 v16h __attribute__((ext_vector_type(16)));
typedef float    v4f  __attribute__((ext_vector_type(4)));
typedef float    v8f  __attribute__((ext_vector_type(8)));

// ---------------- utility kernels ----------------

__global__ void k_fill_f32(float* __restrict__ p, float v, long n) {
  long i = blockIdx.x * (long)blockDim.x + threadIdx.x;
  if (i < n) p[i] = v;
}

__global__ void k_f32_to_f16(const float* __restrict__ s, _Float16* __restrict__ d, long n) {
  long i = blockIdx.x * (long)blockDim.x + threadIdx.x;
  if (i < n) d[i] = (_Float16)s[i];
}

// Ws[l][k][n] (row-major) -> Wt[l][n][k] in f16 (n-major for WMMA B-fragment loads)
__global__ void k_transpose_w(const float* __restrict__ Ws, _Float16* __restrict__ Wt,
                              int H, long total) {
  long i = blockIdx.x * (long)blockDim.x + threadIdx.x;
  if (i >= total) return;
  long hh  = (long)H * H;
  long l   = i / hh;
  long rem = i % hh;
  int  k   = (int)(rem / H);
  int  n   = (int)(rem % H);
  Wt[l * hh + (long)n * H + k] = (_Float16)Ws[i];
}

// deg initialized to 1.0 (self loop); add 1 per incoming edge
__global__ void k_deg(const int* __restrict__ dst, float* __restrict__ deg, int E) {
  long i = blockIdx.x * (long)blockDim.x + threadIdx.x;
  if (i < E) unsafeAtomicAdd(&deg[dst[i]], 1.0f);
}

__global__ void k_rsqrt_inplace(float* __restrict__ p, long n) {
  long i = blockIdx.x * (long)blockDim.x + threadIdx.x;
  if (i < n) p[i] = rsqrtf(p[i]);   // deg >= 1 always
}

__global__ void k_cnts(const int* __restrict__ batch, float* __restrict__ cnts, int N) {
  long i = blockIdx.x * (long)blockDim.x + threadIdx.x;
  if (i < N) unsafeAtomicAdd(&cnts[batch[i]], 1.0f);
}

// ---------------- WMMA GEMM: XWh[N,128] = Xh[N,128] @ W[128,128] ----------------
// One wave per 16-row stripe. B taken from pre-transposed Wt[n][k].
__global__ void gcn_gemm_wmma(const _Float16* __restrict__ Xh,
                              const _Float16* __restrict__ Wt,
                              _Float16* __restrict__ XWh,
                              int nTiles, int N) {
  const int lane = threadIdx.x & 31;
  const int tile = blockIdx.x * (blockDim.x >> 5) + (threadIdx.x >> 5);
  if (tile >= nTiles) return;
  const int m0   = tile << 4;
  const int mrow = lane & 15;
  const bool hi  = (lane & 16) != 0;
  const int kbA  = hi ? 16 : 0;   // A: lanes 16-31 hold K 16..31
  const int kbB  = hi ? 16 : 0;   // B: lanes 16-31 hold K 16..31
  const int ncol = lane & 15;

  v8f acc[8];
  #pragma unroll
  for (int i = 0; i < 8; ++i)
    #pragma unroll
    for (int j = 0; j < 8; ++j) acc[i][j] = 0.0f;

  int arow = m0 + mrow; if (arow >= N) arow = N - 1;   // tail clamp (reads only)

  #pragma unroll
  for (int kt = 0; kt < 4; ++kt) {
    const _Float16* ap = Xh + (size_t)arow * 128 + kt * 32;
    h8v alo = *(const h8v*)(ap + kbA);        // K = kbA .. kbA+7
    h8v ahi = *(const h8v*)(ap + kbA + 8);    // K = kbA+8 .. kbA+15
    v16h a;
    #pragma unroll
    for (int i = 0; i < 8; ++i) { a[i] = alo[i]; a[i + 8] = ahi[i]; }

    #pragma unroll
    for (int nt = 0; nt < 8; ++nt) {
      const _Float16* bp = Wt + (size_t)(nt * 16 + ncol) * 128 + kt * 32 + kbB;
      v16h b = *(const v16h*)bp;              // 16 contiguous K halves
      acc[nt] = __builtin_amdgcn_wmma_f32_16x16x32_f16(
          /*neg_a=*/false, a, /*neg_b=*/false, b,
          /*c_mod=*/(short)0, acc[nt], /*reuse_a=*/false, /*reuse_b=*/false);
    }
  }

  const int mBase = m0 + (hi ? 8 : 0);        // D layout: VGPR r -> M = r (+8 for hi lanes)
  #pragma unroll
  for (int nt = 0; nt < 8; ++nt) {
    #pragma unroll
    for (int r = 0; r < 8; ++r) {
      int m = mBase + r;
      if (m < N) XWh[(size_t)m * 128 + nt * 16 + ncol] = (_Float16)acc[nt][r];
    }
  }
}

// ---------------- edge scatter: acc[dst] += dis[src]*dis[dst] * XWh[src] ----------------
// 32 lanes per edge, 4 features per lane (H=128).
__global__ void gcn_scatter(const int* __restrict__ src, const int* __restrict__ dst,
                            const float* __restrict__ dis,
                            const _Float16* __restrict__ XWh,
                            float* __restrict__ acc, int E) {
  long tid = blockIdx.x * (long)blockDim.x + threadIdx.x;
  long e = tid >> 5;
  if (e >= E) return;
  int f = (int)(tid & 31) * 4;
  int s = src[e], d = dst[e];
  float nrm = dis[s] * dis[d];
  h4v xv = *(const h4v*)(XWh + (size_t)s * 128 + f);   // 8B coalesced gather
  float* op = acc + (size_t)d * 128 + f;
  unsafeAtomicAdd(op + 0, nrm * (float)xv[0]);
  unsafeAtomicAdd(op + 1, nrm * (float)xv[1]);
  unsafeAtomicAdd(op + 2, nrm * (float)xv[2]);
  unsafeAtomicAdd(op + 3, nrm * (float)xv[3]);
}

// ---------------- self-loop + bias + ReLU; emit f16 for next layer, keep f32 in acc ----
__global__ void gcn_bias_relu(float* __restrict__ acc,
                              const _Float16* __restrict__ XWh,
                              const float* __restrict__ dis,
                              const float* __restrict__ bias,
                              _Float16* __restrict__ Xh,
                              int H, long n4) {
  long t = blockIdx.x * (long)blockDim.x + threadIdx.x;
  if (t >= n4) return;
  long base = t * 4;
  long node = base / H;
  int  f    = (int)(base % H);
  float ds  = dis[node];
  float ds2 = ds * ds;                       // self-loop norm
  v4f a  = *(v4f*)(acc + base);
  h4v xw = *(const h4v*)(XWh + base);
  v4f r; h4v rh;
  #pragma unroll
  for (int i = 0; i < 4; ++i) {
    float v = a[i] + ds2 * (float)xw[i] + bias[f + i];
    v = v > 0.0f ? v : 0.0f;
    r[i] = v; rh[i] = (_Float16)v;
  }
  *(v4f*)(acc + base) = r;
  *(h4v*)(Xh + base)  = rh;
}

// ---------------- pooling + head ----------------
__global__ void k_pool(const float* __restrict__ xf, const int* __restrict__ batch,
                       float* __restrict__ pooled, int H, long n4) {
  long t = blockIdx.x * (long)blockDim.x + threadIdx.x;
  if (t >= n4) return;
  long base = t * 4;
  long node = base / H;
  int  f    = (int)(base % H);
  int  g    = batch[node];
  v4f v = *(const v4f*)(xf + base);
  float* pp = pooled + (size_t)g * H + f;
  unsafeAtomicAdd(pp + 0, v[0]);
  unsafeAtomicAdd(pp + 1, v[1]);
  unsafeAtomicAdd(pp + 2, v[2]);
  unsafeAtomicAdd(pp + 3, v[3]);
}

__global__ void k_final(const float* __restrict__ pooled, const float* __restrict__ cnts,
                        const float* __restrict__ Wout, const float* __restrict__ bout,
                        float* __restrict__ out, int G, int C, int H) {
  int i = blockIdx.x * blockDim.x + threadIdx.x;
  if (i >= G * C) return;
  int g = i / C, c = i % C;
  const float* pr = pooled + (size_t)g * H;
  float s = 0.0f;
  for (int h = 0; h < H; ++h) s += pr[h] * Wout[(size_t)h * C + c];
  float cnt = fmaxf(cnts[g], 1.0f);
  out[i] = s / cnt + bout[c];
}

// ---------------- launcher ----------------
extern "C" void kernel_launch(void* const* d_in, const int* in_sizes, int n_in,
                              void* d_out, int out_size, void* d_ws, size_t ws_size,
                              hipStream_t stream) {
  const float* x     = (const float*)d_in[0];
  const int*   ei    = (const int*)d_in[1];
  const int*   batch = (const int*)d_in[2];
  const float* Ws    = (const float*)d_in[4];
  const float* bs    = (const float*)d_in[5];
  const float* Wout  = (const float*)d_in[6];
  const float* bout  = (const float*)d_in[7];
  float* out = (float*)d_out;

  const long NH = in_sizes[0];
  const int  E  = in_sizes[1] / 2;
  const int  N  = in_sizes[2];
  const int  H  = (int)(NH / N);                 // 128
  const int  L  = (int)(in_sizes[4] / (long)(H * H));
  const int  C  = in_sizes[7];
  const int  G  = out_size / C;

  const int* esrc = ei;
  const int* edst = ei + E;

  // workspace carve (256B aligned regions)
  char* wp = (char*)d_ws;
  auto carve = [&](size_t bytes) { void* r = wp; wp += (bytes + 255) & ~(size_t)255; return r; };
  float*    dis    = (float*)carve(sizeof(float) * (size_t)N);
  _Float16* Xh     = (_Float16*)carve(sizeof(_Float16) * (size_t)NH);
  _Float16* XWh    = (_Float16*)carve(sizeof(_Float16) * (size_t)NH);
  float*    acc    = (float*)carve(sizeof(float) * (size_t)NH);
  _Float16* Wt     = (_Float16*)carve(sizeof(_Float16) * (size_t)L * H * H);
  float*    pooled = (float*)carve(sizeof(float) * (size_t)G * H);
  float*    cnts   = (float*)carve(sizeof(float) * (size_t)G);

  const int B = 256;
  auto cdiv = [](long a, long b) { return (int)((a + b - 1) / b); };

  // degree -> dis = rsqrt(deg)  (deg starts at 1.0 for self loop)
  k_fill_f32<<<cdiv(N, B), B, 0, stream>>>(dis, 1.0f, N);
  k_deg<<<cdiv(E, B), B, 0, stream>>>(edst, dis, E);
  k_rsqrt_inplace<<<cdiv(N, B), B, 0, stream>>>(dis, N);

  // precision conversions
  k_f32_to_f16<<<cdiv(NH, B), B, 0, stream>>>(x, Xh, NH);
  long LHH = (long)L * H * H;
  k_transpose_w<<<cdiv(LHH, B), B, 0, stream>>>(Ws, Wt, H, LHH);

  // per-graph node counts
  k_fill_f32<<<cdiv(G, B), B, 0, stream>>>(cnts, 0.0f, G);
  k_cnts<<<cdiv(N, B), B, 0, stream>>>(batch, cnts, N);

  const int nTiles = (N + 15) / 16;
  const long n4 = NH / 4;
  for (int l = 0; l < L; ++l) {
    gcn_gemm_wmma<<<cdiv(nTiles, B / 32), B, 0, stream>>>(Xh, Wt + (size_t)l * H * H, XWh, nTiles, N);
    k_fill_f32<<<cdiv(NH, B), B, 0, stream>>>(acc, 0.0f, NH);
    gcn_scatter<<<cdiv((long)E * 32, B), B, 0, stream>>>(esrc, edst, dis, XWh, acc, E);
    gcn_bias_relu<<<cdiv(n4, B), B, 0, stream>>>(acc, XWh, dis, bs + (size_t)l * H, Xh, H, n4);
  }

  // global mean pool + linear head
  k_fill_f32<<<cdiv((long)G * H, B), B, 0, stream>>>(pooled, 0.0f, (long)G * H);
  k_pool<<<cdiv(n4, B), B, 0, stream>>>(acc, batch, pooled, H, n4);
  k_final<<<cdiv((long)G * C, B), B, 0, stream>>>(pooled, cnts, Wout, bout, out, G, C, H);
}
